// MultiheadAttention_29437705847162
// MI455X (gfx1250) — matrix operand order repaired and checked by reading, not compiled, not run on hardware
//
#include <hip/hip_runtime.h>
#include <hip/hip_bf16.h>

// ---------------------------------------------------------------------------
// Transformer block for MI455X (gfx1250, wave32, WMMA).
//   B=16 S=1024 D=512 H=8 DK=DV=64 OUT=512   => ~95 GFLOP, compute bound.
// GEMMs: V_WMMA_F32_16X16X32_F16, 32x32 supertile per wave, weights pre-packed
// to f16 B-fragment layout (one 32B vector load per fragment).
// Attention: flash-style online softmax, V pre-packed for vector B-frag loads.
// LayerNorm: rows staged to LDS with GLOBAL_LOAD_ASYNC_TO_LDS_B128 (ASYNCcnt).
// ---------------------------------------------------------------------------

typedef __attribute__((ext_vector_type(16))) _Float16 v16h;
typedef __attribute__((ext_vector_type(8)))  _Float16 v8h;
typedef __attribute__((ext_vector_type(8)))  float    v8f;

constexpr int Bsz = 16, Ssz = 1024, Dsz = 512, Hn = 8, DKsz = 64, DVsz = 64, OUTsz = 512;
constexpr int BS  = Bsz * Ssz;                   // 16384 rows
constexpr float NEGV    = -4294967295.0f;        // mask fill value (matches reference)
constexpr float INVTEMP = 1.0f / (8.0f + 1e-6f); // 1/(sqrt(64)+eps)

#define WMMA_F16(a, b, c) \
  __builtin_amdgcn_wmma_f32_16x16x32_f16(false, (a), false, (b), (short)0, (c), false, false)

// ---------------------------------------------------------------------------
// Pack an f32 weight matrix into f16 B-fragment tiles.
// Fragment tile (kt, nt): lane holds column n = nt*16 + (lane&15); its 16
// halves are k = kt*32 + (lane>>4)*16 + i  (contiguous k order, same as the A loads).
// Packed address: ((kt*tilesN + nt)*32 + lane)*16 + i   (32B/lane, aligned).
// ---------------------------------------------------------------------------
template <bool TRANS>
__global__ void pack_weight(const float* __restrict__ W, _Float16* __restrict__ P,
                            int K, int N) {
  const int idx  = blockIdx.x * blockDim.x + threadIdx.x;
  const int lane = idx & 31;
  const int tile = idx >> 5;
  const int tilesN = N >> 4;
  if (tile >= (K >> 5) * tilesN) return;
  const int kt = tile / tilesN, nt = tile % tilesN;
  const int n     = nt * 16 + (lane & 15);
  const int kbase = kt * 32 + (lane >> 4) * 16;
  v16h val;
#pragma unroll
  for (int i = 0; i < 16; ++i) {
    const float w = TRANS ? W[(size_t)n * K + kbase + i] : W[(size_t)(kbase + i) * N + n];
    val[i] = (_Float16)w;
  }
  *(v16h*)(P + ((size_t)tile * 32 + lane) * 16) = val;
}

// ---------------------------------------------------------------------------
// GEMM: C[M,N] = act(A[M,K] * B + bias).  One wave computes a 32x32 supertile
// (2x2 C tiles, 4 WMMAs per 32-wide k-step).  B comes pre-packed (f16).
// OUT_MODE: 0 f32 [M,N]; 1 f16 [M,N]; 2 f16 heads [b,h,s,dk]; 3 f16 V-packed.
// ---------------------------------------------------------------------------
template <bool A_F16, bool BIAS, bool RELU, int OUT_MODE>
__global__ void gemm_wmma(const void* __restrict__ Aab, const _Float16* __restrict__ Bp,
                          const float* __restrict__ bias, void* __restrict__ Out,
                          int M, int N, int K) {
  const int lane = threadIdx.x & 31;
  const int warp = threadIdx.x >> 5;
  const int grp  = lane >> 4;
  const int l16  = lane & 15;

  const int tilesN2 = N >> 5;                    // 32-wide col supertiles
  const int tiles   = (M >> 5) * tilesN2;
  const int wg      = blockIdx.x * (blockDim.x >> 5) + warp;
  if (wg >= tiles) return;
  const int tm = wg / tilesN2;
  const int tn = wg % tilesN2;
  const int tilesN = N >> 4;

  v8f acc[2][2] = {};
  for (int k0 = 0; k0 < K; k0 += 32) {
    const int kt = k0 >> 5;
    // ---- two A fragments (rows tm*32 + r*16 + l16); contiguous 8-half runs
    v16h af[2];
#pragma unroll
    for (int r = 0; r < 2; ++r) {
      const int arow = tm * 32 + r * 16 + l16;
      if (A_F16) {
        const _Float16* rp = (const _Float16*)Aab + (size_t)arow * K + k0;
        v8h lo = *(const v8h*)(rp + grp * 16);
        v8h hi = *(const v8h*)(rp + grp * 16 + 8);
#pragma unroll
        for (int i = 0; i < 8; ++i) { af[r][i] = lo[i]; af[r][8 + i] = hi[i]; }
      } else {
        const float* rp = (const float*)Aab + (size_t)arow * K + k0;
        __builtin_prefetch(rp + 32, 0, 1);
#pragma unroll
        for (int i = 0; i < 8; ++i) af[r][i] = (_Float16)rp[grp * 16 + i];
#pragma unroll
        for (int i = 0; i < 8; ++i) af[r][8 + i] = (_Float16)rp[grp * 16 + 8 + i];
      }
    }
    // ---- two B fragments: single 32B vector load each from packed weights
    v16h bf[2];
#pragma unroll
    for (int c = 0; c < 2; ++c)
      bf[c] = *(const v16h*)(Bp + (((size_t)kt * tilesN + (tn * 2 + c)) * 32 + lane) * 16);

#pragma unroll
    for (int r = 0; r < 2; ++r)
#pragma unroll
      for (int c = 0; c < 2; ++c) acc[r][c] = WMMA_F16(af[r], bf[c], acc[r][c]);
  }

  // ---- epilogue: C rows M = v + 8*grp within each tile
#pragma unroll
  for (int r = 0; r < 2; ++r)
#pragma unroll
    for (int c = 0; c < 2; ++c) {
      const int n  = tn * 32 + c * 16 + l16;
      const float bv = BIAS ? bias[n] : 0.0f;
#pragma unroll
      for (int v = 0; v < 8; ++v) {
        float x = acc[r][c][v] + bv;
        if (RELU) x = x > 0.0f ? x : 0.0f;
        const int m = tm * 32 + r * 16 + grp * 8 + v;
        if (OUT_MODE == 0) {
          ((float*)Out)[(size_t)m * N + n] = x;
        } else if (OUT_MODE == 1) {
          ((_Float16*)Out)[(size_t)m * N + n] = (_Float16)x;
        } else if (OUT_MODE == 2) {  // heads: [b,h,s,dk] f16 (Q and K)
          const int b = m / Ssz, s = m % Ssz;
          const int h = n / DKsz, dk = n % DKsz;
          ((_Float16*)Out)[(((size_t)b * Hn + h) * Ssz + s) * DKsz + dk] = (_Float16)x;
        } else {                      // V packed into attention B-fragment layout
          const int b = m / Ssz, s = m % Ssz;
          const int h = n / DVsz, dv = n % DVsz;
          const int bh = b * Hn + h;
          const int sc = s >> 5, w = s & 31;
          const int lanev = (w >> 4) * 16 + (dv & 15);
          const int i = w & 15;
          const int t = dv >> 4;
          ((_Float16*)Out)[((((size_t)bh * (Ssz / 32) + sc) * 4 + t) * 32 + lanev) * 16 + i] =
              (_Float16)x;
        }
      }
    }
}

// ---------------------------------------------------------------------------
// Flash attention: one wave owns a 16-query tile of one (b,h); streams K/V in
// 32-key chunks. Scores via 4 WMMAs (K rows load as contiguous v16h), online
// softmax (shuffle reductions within 16-lane halves), P reshaped through LDS,
// P*V via 4 WMMAs with vector B-frag loads from packed V.
// ---------------------------------------------------------------------------
__global__ void attn_kernel(const _Float16* __restrict__ Qh, const _Float16* __restrict__ Kh,
                            const _Float16* __restrict__ Vp, const unsigned char* __restrict__ msk,
                            _Float16* __restrict__ Oh) {
  __shared__ _Float16 probs[8][16][32];

  const int lane = threadIdx.x & 31;
  const int warp = threadIdx.x >> 5;
  const int grp  = lane >> 4;
  const int l16  = lane & 15;

  const int wg = blockIdx.x * 8 + warp;          // B*H*(S/16) = 8192 waves
  const int qt = wg % (Ssz / 16);
  const int bh = wg / (Ssz / 16);
  const int b  = bh / Hn;
  const int h  = bh % Hn;

  const _Float16* Qb = Qh + (size_t)bh * Ssz * DKsz;
  const _Float16* Kb = Kh + (size_t)bh * Ssz * DKsz;
  const _Float16* Vb = Vp + (size_t)bh * (Ssz / 32) * 4 * 32 * 16;

  // resident Q fragments: qa[t] covers dk in [32t, 32t+32)
  v16h qa[2];
  {
    const _Float16* qrow = Qb + (size_t)(qt * 16 + l16) * DKsz;
#pragma unroll
    for (int t = 0; t < 2; ++t) {
      v8h lo = *(const v8h*)(qrow + t * 32 + grp * 16);
      v8h hi = *(const v8h*)(qrow + t * 32 + grp * 16 + 8);
#pragma unroll
      for (int i = 0; i < 8; ++i) { qa[t][i] = lo[i]; qa[t][8 + i] = hi[i]; }
    }
  }

  float mrow[8], lrow[8];
#pragma unroll
  for (int v = 0; v < 8; ++v) { mrow[v] = -3.0e38f; lrow[v] = 0.0f; }
  v8f o[4] = {};

  for (int kc = 0; kc < Ssz; kc += 32) {
    // ---- scores: two 16x16 tiles (keys kc..+15 and kc+16..+31)
    v8f s0 = {}, s1 = {};
#pragma unroll
    for (int ks = 0; ks < 2; ++ks) {
      v16h bk0 = *(const v16h*)(Kb + (size_t)(kc + l16) * DKsz + ks * 32 + grp * 16);
      v16h bk1 = *(const v16h*)(Kb + (size_t)(kc + 16 + l16) * DKsz + ks * 32 + grp * 16);
      s0 = WMMA_F16(qa[ks], bk0, s0);
      s1 = WMMA_F16(qa[ks], bk1, s1);
    }

    const bool m0 = msk[(size_t)b * Ssz + kc + l16] != 0;
    const bool m1 = msk[(size_t)b * Ssz + kc + 16 + l16] != 0;

    // ---- online softmax over the 32-key chunk
    float p0[8], p1[8], sc[8];
#pragma unroll
    for (int v = 0; v < 8; ++v) {
      float x0 = m0 ? NEGV : s0[v] * INVTEMP;
      float x1 = m1 ? NEGV : s1[v] * INVTEMP;
      float t = fmaxf(x0, x1);
#pragma unroll
      for (int d = 1; d < 16; d <<= 1) t = fmaxf(t, __shfl_xor(t, d, 32));
      const float mn = fmaxf(mrow[v], t);
      const float scale = __expf(mrow[v] - mn);
      const float e0 = __expf(x0 - mn), e1 = __expf(x1 - mn);
      float rs = e0 + e1;
#pragma unroll
      for (int d = 1; d < 16; d <<= 1) rs += __shfl_xor(rs, d, 32);
      lrow[v] = lrow[v] * scale + rs;
      mrow[v] = mn;
      sc[v] = scale; p0[v] = e0; p1[v] = e1;
    }
#pragma unroll
    for (int t = 0; t < 4; ++t)
#pragma unroll
      for (int v = 0; v < 8; ++v) o[t][v] *= sc[v];

    // ---- reshape P: C-layout -> LDS -> A-layout (same-wave DS ops in-order)
#pragma unroll
    for (int v = 0; v < 8; ++v) {
      probs[warp][v + grp * 8][l16]      = (_Float16)p0[v];
      probs[warp][v + grp * 8][16 + l16] = (_Float16)p1[v];
    }
    v16h pa;
    {
      const _Float16* pr = &probs[warp][l16][0];
#pragma unroll
      for (int i = 0; i < 8; ++i) pa[i] = pr[grp * 16 + i];
#pragma unroll
      for (int i = 0; i < 8; ++i) pa[8 + i] = pr[grp * 16 + 8 + i];
    }

    // ---- o[t] += P(16x32) * V(32x16): vector B-frag loads from packed V
    const int scIdx = kc >> 5;
#pragma unroll
    for (int t = 0; t < 4; ++t) {
      v16h vb = *(const v16h*)(Vb + (((size_t)scIdx * 4 + t) * 32 + lane) * 16);
      o[t] = WMMA_F16(pa, vb, o[t]);
    }
  }

  // ---- normalize and store as f16 [b, s, h*64 + n]
#pragma unroll
  for (int t = 0; t < 4; ++t)
#pragma unroll
    for (int v = 0; v < 8; ++v) {
      const float val = o[t][v] / lrow[v];
      const int s = qt * 16 + v + grp * 8;
      Oh[((size_t)b * Ssz + s) * (Hn * DVsz) + h * DVsz + t * 16 + l16] = (_Float16)val;
    }
}

// ---------------------------------------------------------------------------
// Row-wise residual + LayerNorm: out_f16 = LN(Y + Res) * g + b.  D = 512.
// Y row is staged to LDS with the CDNA5 async-copy path (ASYNCcnt tracked).
// ---------------------------------------------------------------------------
template <bool RES_F16>
__global__ void ln_residual_kernel(const float* __restrict__ Y, const void* __restrict__ Res,
                                   const float* __restrict__ g, const float* __restrict__ bt,
                                   _Float16* __restrict__ Out) {
  __shared__ __align__(16) float yrow[Dsz];
  __shared__ float red[256];
  const int row = blockIdx.x;
  const int tid = threadIdx.x;
  const int i0 = tid, i1 = tid + 256;

  // ---- async stage: 128 lanes x 16B = 512 floats, memory -> LDS
  if (tid < 128) {
    const float* gp = Y + (size_t)row * Dsz + tid * 4;
    const unsigned int lds = (unsigned int)(size_t)(&yrow[tid * 4]);
    asm volatile("global_load_async_to_lds_b128 %0, %1, off"
                 :: "v"(lds), "v"(gp)
                 : "memory");
  }
  asm volatile("s_wait_asynccnt 0x0" ::: "memory");
  __syncthreads();

  float r0, r1;
  if (RES_F16) {
    const _Float16* R = (const _Float16*)Res;
    r0 = (float)R[(size_t)row * Dsz + i0];
    r1 = (float)R[(size_t)row * Dsz + i1];
  } else {
    const float* R = (const float*)Res;
    r0 = R[(size_t)row * Dsz + i0];
    r1 = R[(size_t)row * Dsz + i1];
  }
  const float x0 = yrow[i0] + r0;
  const float x1 = yrow[i1] + r1;

  red[tid] = x0 + x1;
  __syncthreads();
  for (int d = 128; d > 0; d >>= 1) { if (tid < d) red[tid] += red[tid + d]; __syncthreads(); }
  const float mu = red[0] * (1.0f / (float)Dsz);
  __syncthreads();

  const float d0 = x0 - mu, d1 = x1 - mu;
  red[tid] = d0 * d0 + d1 * d1;
  __syncthreads();
  for (int d = 128; d > 0; d >>= 1) { if (tid < d) red[tid] += red[tid + d]; __syncthreads(); }
  const float rstd = rsqrtf(red[0] * (1.0f / (float)Dsz) + 1e-5f);

  Out[(size_t)row * Dsz + i0] = (_Float16)(d0 * rstd * g[i0] + bt[i0]);
  Out[(size_t)row * Dsz + i1] = (_Float16)(d1 * rstd * g[i1] + bt[i1]);
}

// ---------------------------------------------------------------------------
extern "C" void kernel_launch(void* const* d_in, const int* in_sizes, int n_in,
                              void* d_out, int out_size, void* d_ws, size_t ws_size,
                              hipStream_t stream) {
  (void)in_sizes; (void)n_in; (void)out_size; (void)ws_size;
  const float* query = (const float*)d_in[0];
  const float* key   = (const float*)d_in[1];
  const float* value = (const float*)d_in[2];
  const unsigned char* mask = (const unsigned char*)d_in[3];
  const float* W_q = (const float*)d_in[4];
  const float* W_k = (const float*)d_in[5];
  const float* W_v = (const float*)d_in[6];
  const float* W_o = (const float*)d_in[7];
  const float* w1  = (const float*)d_in[8];
  const float* b1  = (const float*)d_in[9];
  const float* w2  = (const float*)d_in[10];
  const float* b2  = (const float*)d_in[11];
  const float* w3  = (const float*)d_in[12];
  const float* b3  = (const float*)d_in[13];
  const float* lng = (const float*)d_in[14];
  const float* lnb = (const float*)d_in[15];
  float* out = (float*)d_out;
  char*  ws  = (char*)d_ws;

  // Workspace layout: activations (f16 16MB each / f32 32MB) + packed weights.
  constexpr size_t F16M = (size_t)BS * Dsz * sizeof(_Float16);   // 16 MB
  constexpr size_t WPK  = (size_t)Dsz * Dsz * sizeof(_Float16);  // 512 KB
  _Float16* Qh  = (_Float16*)(ws);
  _Float16* Khm = (_Float16*)(ws + 1 * F16M);
  _Float16* Vpk = (_Float16*)(ws + 2 * F16M);
  _Float16* Ath = (_Float16*)(ws + 3 * F16M);
  float*    Yf  = (float*)   (ws + 4 * F16M);                       // 32 MB
  _Float16* X1h = (_Float16*)(ws + 4 * F16M + (size_t)BS * Dsz * sizeof(float));
  char* wbase = ws + 5 * F16M + (size_t)BS * Dsz * sizeof(float);
  _Float16* Wq_p = (_Float16*)(wbase + 0 * WPK);
  _Float16* Wk_p = (_Float16*)(wbase + 1 * WPK);
  _Float16* Wv_p = (_Float16*)(wbase + 2 * WPK);
  _Float16* Wo_p = (_Float16*)(wbase + 3 * WPK);
  _Float16* w1_p = (_Float16*)(wbase + 4 * WPK);
  _Float16* w2_p = (_Float16*)(wbase + 5 * WPK);
  _Float16* w3_p = (_Float16*)(wbase + 6 * WPK);
  _Float16* H1h = Qh;    // reuse: Q dead after attention
  _Float16* X2h = Khm;   // reuse: K dead after attention

  const dim3 blk(256);
  const int packBlocks = (Dsz / 32) * (Dsz / 16) * 32 / 256;   // 64 blocks
  const int gemmBlocks = (BS / 32) * (Dsz / 32) / 8;           // 1024 blocks, 8 waves
  const int attnBlocks = Bsz * Hn * (Ssz / 16) / 8;            // 1024 blocks

  // 0) pack weights to f16 B-fragment layout (W_q..W_o normal, w1..w3 as W^T)
  pack_weight<false><<<packBlocks, blk, 0, stream>>>(W_q, Wq_p, Dsz, Dsz);
  pack_weight<false><<<packBlocks, blk, 0, stream>>>(W_k, Wk_p, Dsz, Dsz);
  pack_weight<false><<<packBlocks, blk, 0, stream>>>(W_v, Wv_p, Dsz, Dsz);
  pack_weight<false><<<packBlocks, blk, 0, stream>>>(W_o, Wo_p, Dsz, Dsz);
  pack_weight<true ><<<packBlocks, blk, 0, stream>>>(w1,  w1_p, Dsz, Dsz);
  pack_weight<true ><<<packBlocks, blk, 0, stream>>>(w2,  w2_p, Dsz, Dsz);
  pack_weight<true ><<<packBlocks, blk, 0, stream>>>(w3,  w3_p, Dsz, Dsz);

  // 1) QKV projections (f32 activations in; Q/K head layout, V attention-packed)
  gemm_wmma<false, false, false, 2><<<gemmBlocks, blk, 0, stream>>>(query, Wq_p, nullptr, Qh,  BS, Dsz, Dsz);
  gemm_wmma<false, false, false, 2><<<gemmBlocks, blk, 0, stream>>>(key,   Wk_p, nullptr, Khm, BS, Dsz, Dsz);
  gemm_wmma<false, false, false, 3><<<gemmBlocks, blk, 0, stream>>>(value, Wv_p, nullptr, Vpk, BS, Dsz, Dsz);

  // 2) flash attention -> f16 [b, s, h*dv]
  attn_kernel<<<attnBlocks, blk, 0, stream>>>(Qh, Khm, Vpk, mask, Ath);

  // 3) output projection, then residual+LN with query
  gemm_wmma<true, false, false, 0><<<gemmBlocks, blk, 0, stream>>>(Ath, Wo_p, nullptr, Yf, BS, Dsz, Dsz);
  ln_residual_kernel<false><<<BS, 256, 0, stream>>>(Yf, query, lng, lnb, X1h);

  // 4) FFN: relu(X1 w1^T + b1) w2^T + b2, then residual+LN with X1
  gemm_wmma<true, true, true,  1><<<gemmBlocks, blk, 0, stream>>>(X1h, w1_p, b1, H1h, BS, Dsz, Dsz);
  gemm_wmma<true, true, false, 0><<<gemmBlocks, blk, 0, stream>>>(H1h, w2_p, b2, Yf,  BS, Dsz, Dsz);
  ln_residual_kernel<true><<<BS, 256, 0, stream>>>(Yf, X1h, lng, lnb, X2h);

  // 5) final projection: X2 w3^T + b3 -> f32 output
  gemm_wmma<true, true, false, 0><<<gemmBlocks, blk, 0, stream>>>(X2h, w3_p, b3, out, BS, OUTsz, Dsz);
}
